// SyntaxGatedAttention_2637109920540
// MI455X (gfx1250) — matrix operand order repaired and checked by reading, not compiled, not run on hardware
//
#include <hip/hip_runtime.h>
#include <hip/hip_bf16.h>

typedef __attribute__((ext_vector_type(16))) _Float16     v16h;
typedef __attribute__((ext_vector_type(8)))  _Float16     v8h;
typedef __attribute__((ext_vector_type(8)))  float        v8f;
typedef __attribute__((ext_vector_type(4)))  unsigned int u32x4;
typedef __attribute__((ext_vector_type(8)))  unsigned int u32x8;

#define WMMA_F16(a, b, c) \
  __builtin_amdgcn_wmma_f32_16x16x32_f16(false, (a), false, (b), (short)0, (c), false, false)

static __device__ inline v16h cat8(v8h lo, v8h hi) {
  return __builtin_shufflevector(lo, hi, 0, 1, 2, 3, 4, 5, 6, 7, 8, 9, 10, 11, 12, 13, 14, 15);
}

// ---------------------------------------------------------------------------
// TDM: issue a 2-D tensor_load_to_lds of a (tile_d1 x tile_d0) f16 tile from a
// row-major matrix (row stride = ld elements) into LDS at lds_off, padding each
// 32-half line with 8 halves (pad_interval=16 DW, pad_amount=4 DW) to match the
// [..][40] LDS row layout used by the WMMA fragment loads. Wave-level op,
// tracked by TENSORcnt.
// ---------------------------------------------------------------------------
static __device__ inline void tdm_load_tile_f16(unsigned lds_off, const _Float16* gptr,
                                                unsigned tile_d0, unsigned tile_d1,
                                                unsigned dim0, unsigned dim1, unsigned ld) {
  unsigned long long ga = (unsigned long long)(uintptr_t)gptr;
  u32x4 g0;
  g0[0] = 1u;                                    // count=1, user descriptor
  g0[1] = lds_off;                               // lds_addr (bytes)
  g0[2] = (unsigned)(ga & 0xffffffffull);        // global_addr[31:0]
  g0[3] = (unsigned)((ga >> 32) & 0x01ffffffull) // global_addr[56:32]
        | (2u << 30);                            // type = 2 ("image")
  u32x8 g1;
  g1[0] = (1u << 16)                             // data_size = 2 bytes
        | (1u << 20)                             // pad_enable
        | (3u << 22)                             // pad_interval: 16 DWORDs (32 halves)
        | (3u << 25);                            // pad_amount:   4 DWORDs (8 halves)
  g1[1] = (dim0 & 0xffffu) << 16;                // tensor_dim0[15:0]  (bits 63:48)
  g1[2] = ((dim0 >> 16) & 0xffffu)               // tensor_dim0[31:16] (bits 79:64)
        | ((dim1 & 0xffffu) << 16);              // tensor_dim1[15:0]  (bits 95:80)
  g1[3] = ((dim1 >> 16) & 0xffffu)               // tensor_dim1[31:16]
        | (tile_d0 << 16);                       // tile_dim0 (bits 127:112)
  g1[4] = tile_d1 & 0xffffu;                     // tile_dim1 (tile_dim2 = 0)
  g1[5] = ld;                                    // tensor_dim0_stride[31:0]
  g1[6] = 0u;                                    // stride0[47:32], stride1[15:0]
  g1[7] = 0u;
  asm volatile("tensor_load_to_lds %0, %1" ::"s"(g0), "s"(g1) : "memory");
}

// ---------------------------------------------------------------------------
// f32 -> f16 conversion
// ---------------------------------------------------------------------------
__global__ void cvt_f16_kernel(const float* __restrict__ src, _Float16* __restrict__ dst,
                               size_t n) {
  size_t i = (size_t)blockIdx.x * blockDim.x + threadIdx.x;
  if (i < n) dst[i] = (_Float16)src[i];
}

// ---------------------------------------------------------------------------
// Tiled f16 WMMA GEMM: C(MxN) = A(MxK) * B(KxN), row-major, f32 accum.
// Block tile 128x128, 8 waves (2x4) of 64x32, K step 32.
// A tile staged by the Tensor Data Mover (wave 0); B staged manually with a
// transpose so B-fragments are contiguous LDS runs.
// ---------------------------------------------------------------------------
template <bool F32OUT>
__global__ __launch_bounds__(256) void gemm_f16_kernel(
    const _Float16* __restrict__ A, const _Float16* __restrict__ B,
    void* __restrict__ Cout, int M, int N, int K) {
  __shared__ _Float16 As[128][40];  // row-major A tile (40-half rows: TDM padded)
  __shared__ _Float16 Bt[128][40];  // transposed B tile: Bt[n][k]

  const int tid  = threadIdx.x;
  const int lane = tid & 31;
  const int wave = tid >> 5;
  const int wm   = wave >> 2;  // 0..1
  const int wn   = wave & 3;   // 0..3
  const int l16  = lane & 15;
  const int hi   = lane >> 4;
  const int row0 = blockIdx.y * 128;
  const int col0 = blockIdx.x * 128;
  const unsigned as_base = (unsigned)(uintptr_t)&As[0][0];

  const v8f vzero = {0.f, 0.f, 0.f, 0.f, 0.f, 0.f, 0.f, 0.f};
  v8f acc[4][2];
#pragma unroll
  for (int i = 0; i < 4; ++i)
#pragma unroll
    for (int j = 0; j < 2; ++j) acc[i][j] = vzero;

  for (int k0 = 0; k0 < K; k0 += 32) {
    __syncthreads();  // previous iteration's LDS reads complete
    if (wave == 0) {  // TDM: A tile (128 rows x 32 halves), padded to 40-half rows
      tdm_load_tile_f16(as_base, A + (size_t)row0 * K + k0,
                        /*tile_d0=*/32, /*tile_d1=*/128,
                        /*dim0=*/(unsigned)K, /*dim1=*/(unsigned)M, (unsigned)K);
    }
    {  // stage B tile (32 x 128), transposed into Bt[n][k]
      int kk = tid >> 3;        // 0..31
      int ns = (tid & 7) * 16;  // 0..112
      const _Float16* src = B + (size_t)(k0 + kk) * N + col0 + ns;
      v8h b0 = *(const v8h*)src;
      v8h b1 = *(const v8h*)(src + 8);
      if (k0 + 32 < K) __builtin_prefetch(src + (size_t)32 * N, 0, 0);
#pragma unroll
      for (int i = 0; i < 8; ++i) Bt[ns + i][kk] = b0[i];
#pragma unroll
      for (int i = 0; i < 8; ++i) Bt[ns + 8 + i][kk] = b1[i];
    }
    if (wave == 0) __builtin_amdgcn_s_wait_tensorcnt(0);
    __syncthreads();

    v16h bfrag[2];
#pragma unroll
    for (int j = 0; j < 2; ++j) {
      int n  = wn * 32 + j * 16 + l16;  // B-frag: lane = column, K contiguous
      int kf = hi * 8;
      v8h lo  = *(const v8h*)&Bt[n][kf];
      v8h hi2 = *(const v8h*)&Bt[n][kf + 16];
      bfrag[j] = cat8(lo, hi2);
    }
#pragma unroll
    for (int i = 0; i < 4; ++i) {
      int r  = wm * 64 + i * 16 + l16;  // A-frag: lane = row
      int kb = hi * 8;                  // K: kb..kb+7 and 16+kb..16+kb+7
      v8h lo  = *(const v8h*)&As[r][kb];
      v8h hi2 = *(const v8h*)&As[r][kb + 16];
      v16h afrag = cat8(lo, hi2);
#pragma unroll
      for (int j = 0; j < 2; ++j) acc[i][j] = WMMA_F16(afrag, bfrag[j], acc[i][j]);
    }
  }

  // branch-free epilogue, incremental addressing
#pragma unroll
  for (int i = 0; i < 4; ++i)
#pragma unroll
    for (int j = 0; j < 2; ++j) {
      const int c = col0 + wn * 32 + j * 16 + l16;
      size_t idx = (size_t)(row0 + wm * 64 + i * 16 + hi * 8) * N + c;
      if (F32OUT) {
        float* p = (float*)Cout;
#pragma unroll
        for (int v = 0; v < 8; ++v) { p[idx] = acc[i][j][v]; idx += N; }
      } else {
        _Float16* p = (_Float16*)Cout;
#pragma unroll
        for (int v = 0; v < 8; ++v) { p[idx] = (_Float16)acc[i][j][v]; idx += N; }
      }
    }
}

// ---------------------------------------------------------------------------
// blend = sigmoid(x @ Wblend + bblend), (B*T, H) — tiny GEMM, VALU
// ---------------------------------------------------------------------------
__global__ void blend_kernel(const float* __restrict__ x, const float* __restrict__ Wb,
                             const float* __restrict__ bb, float* __restrict__ blend) {
  int idx = blockIdx.x * blockDim.x + threadIdx.x;  // BT*H = 65536
  int t = idx >> 4, h = idx & 15;
  const float* xr = x + (size_t)t * 2048;
  float s = bb[h];
  for (int d = 0; d < 2048; ++d) s += xr[d] * Wb[d * 16 + h];
  blend[idx] = 1.f / (1.f + __expf(-s));
}

// ---------------------------------------------------------------------------
// per-batch mean of is_code_mask
// ---------------------------------------------------------------------------
__global__ void mean_kernel(const float* __restrict__ m, float* __restrict__ ikm, int T) {
  __shared__ float red[256];
  int b = blockIdx.x;
  float s = 0.f;
  for (int t = threadIdx.x; t < T; t += 256) s += m[(size_t)b * T + t];
  red[threadIdx.x] = s;
  __syncthreads();
  for (int w = 128; w > 0; w >>= 1) {
    if (threadIdx.x < (unsigned)w) red[threadIdx.x] += red[threadIdx.x + w];
    __syncthreads();
  }
  if (threadIdx.x == 0) ikm[b] = red[0] / (float)T;
}

// ---------------------------------------------------------------------------
// gate MLP: 2 -> 8 (silu) -> 4 (sigmoid), per token
// ---------------------------------------------------------------------------
__global__ void gate_kernel(const float* __restrict__ mask, const float* __restrict__ ikm,
                            const float* __restrict__ Wg1, const float* __restrict__ bg1,
                            const float* __restrict__ Wg2, const float* __restrict__ bg2,
                            float* __restrict__ gate, int T) {
  int idx = blockIdx.x * blockDim.x + threadIdx.x;  // B*T
  if (idx >= 2 * T) return;
  int b = idx / T;
  float i0 = mask[idx];
  float i1 = ikm[b];
  float hbuf[8];
#pragma unroll
  for (int j = 0; j < 8; ++j) {
    float z = i0 * Wg1[j] + i1 * Wg1[8 + j] + bg1[j];
    hbuf[j] = z / (1.f + __expf(-z));  // silu
  }
#pragma unroll
  for (int g = 0; g < 4; ++g) {
    float z = bg2[g];
#pragma unroll
    for (int j = 0; j < 8; ++j) z += hbuf[j] * Wg2[j * 4 + g];
    gate[idx * 4 + g] = 1.f / (1.f + __expf(-z));
  }
}

// ---------------------------------------------------------------------------
// k/v blend-combine: fp32 outputs (B,H,T,DH) to d_out + f16 copies for attn
// ---------------------------------------------------------------------------
__global__ void combine_kv_kernel(const _Float16* __restrict__ kc, const _Float16* __restrict__ kt,
                                  const _Float16* __restrict__ vc, const _Float16* __restrict__ vt,
                                  const float* __restrict__ blend,
                                  float* __restrict__ kout, float* __restrict__ vout,
                                  _Float16* __restrict__ kh, _Float16* __restrict__ vh) {
  size_t e = (size_t)blockIdx.x * blockDim.x + threadIdx.x;  // B*H*T*DH = 8388608
  if (e >= 8388608ull) return;
  int d = (int)(e & 127);
  int t = (int)((e >> 7) & 2047);
  int h = (int)((e >> 18) & 15);
  int b = (int)(e >> 22);
  size_t src = ((size_t)(b * 2048 + t)) * 2048 + h * 128 + d;  // (B*T, D) projection layout
  float bl  = blend[(size_t)(b * 2048 + t) * 16 + h];
  float kv  = bl * (float)kc[src] + (1.f - bl) * (float)kt[src];
  float vv  = bl * (float)vc[src] + (1.f - bl) * (float)vt[src];
  kout[e] = kv;
  vout[e] = vv;
  kh[e] = (_Float16)kv;
  vh[e] = (_Float16)vv;
}

// ---------------------------------------------------------------------------
// Flash attention: block = (b,h, 128 q rows), 8 waves x 16 q rows each.
// ---------------------------------------------------------------------------
__global__ __launch_bounds__(256) void attn_kernel(
    const _Float16* __restrict__ qh, const _Float16* __restrict__ kh,
    const _Float16* __restrict__ vh, const float* __restrict__ gate,
    _Float16* __restrict__ oh) {
  constexpr int T = 2048, DH = 128, G = 4, D = 2048;
  __shared__ _Float16 Ps[8][16][40];  // per-wave P tile (16 x 32), padded
  __shared__ _Float16 Vt[128][40];    // transposed V chunk: Vt[d][k]

  const int tid = threadIdx.x, lane = tid & 31, wave = tid >> 5;
  const int l16 = lane & 15, hi = lane >> 4;
  const int bh = blockIdx.x;  // b*H + h
  const int b = bh >> 4, h = bh & 15;
  const int q0 = blockIdx.y * 128 + wave * 16;
  const float invScale = 0.08838834764831845f;  // 1/sqrt(128)
  const size_t kvbase = (size_t)bh * T * DH;

  // Q A-fragments: 4 chunks of K=32 across DH
  v16h qa[4];
  {
    const int qrow = q0 + l16;
    const _Float16* qp = qh + ((size_t)(b * T + qrow)) * D + h * DH;
    const int kb = hi * 8;
#pragma unroll
    for (int c = 0; c < 4; ++c) {
      v8h lo  = *(const v8h*)(qp + c * 32 + kb);
      v8h hi2 = *(const v8h*)(qp + c * 32 + 16 + kb);
      qa[c] = cat8(lo, hi2);
    }
  }

  // per-q-row gate factor (heads < G only)
  float gr[8];
#pragma unroll
  for (int v = 0; v < 8; ++v) {
    int r = q0 + v + hi * 8;
    gr[v] = (h < G) ? gate[((size_t)(b * T + r)) * G + h] : 1.f;
  }

  const v8f vzero = {0.f, 0.f, 0.f, 0.f, 0.f, 0.f, 0.f, 0.f};
  v8f od[8];
  float mrow[8], lrow[8];
#pragma unroll
  for (int i = 0; i < 8; ++i) { od[i] = vzero; mrow[i] = -1e30f; lrow[i] = 0.f; }

  for (int k0 = 0; k0 < T; k0 += 32) {
    __syncthreads();
    {  // cooperatively stage V chunk (32 x 128) transposed
      int kk = tid >> 3;        // 0..31
      int ds = (tid & 7) * 16;  // 0..112
      const _Float16* vp = vh + kvbase + (size_t)(k0 + kk) * DH + ds;
      v8h a0 = *(const v8h*)vp;
      v8h a1 = *(const v8h*)(vp + 8);
      if (k0 + 32 < T) __builtin_prefetch(vp + (size_t)32 * DH, 0, 0);
#pragma unroll
      for (int i = 0; i < 8; ++i) Vt[ds + i][kk] = a0[i];
#pragma unroll
      for (int i = 0; i < 8; ++i) Vt[ds + 8 + i][kk] = a1[i];
    }

    // S = Q K^T (16q x 32k): two 16x16 tiles, contraction over DH
    v8f s0 = vzero, s1 = vzero;
#pragma unroll
    for (int c = 0; c < 4; ++c) {
      int kf = hi * 8;
      const _Float16* kp0 = kh + kvbase + (size_t)(k0 + l16) * DH + c * 32 + kf;
      const _Float16* kp1 = kh + kvbase + (size_t)(k0 + 16 + l16) * DH + c * 32 + kf;
      v16h kb0 = cat8(*(const v8h*)kp0, *(const v8h*)(kp0 + 16));
      v16h kb1 = cat8(*(const v8h*)kp1, *(const v8h*)(kp1 + 16));
      s0 = WMMA_F16(qa[c], kb0, s0);
      s1 = WMMA_F16(qa[c], kb1, s1);
    }

    // scale + per-row gate
#pragma unroll
    for (int v = 0; v < 8; ++v) {
      float g = invScale * gr[v];
      s0[v] *= g;
      s1[v] *= g;
    }

    // online softmax (row reductions within 16-lane halves)
    float fs[8];
#pragma unroll
    for (int v = 0; v < 8; ++v) {
      float mx = fmaxf(s0[v], s1[v]);
#pragma unroll
      for (int m = 8; m > 0; m >>= 1) mx = fmaxf(mx, __shfl_xor(mx, m, 32));
      float mnew = fmaxf(mrow[v], mx);
      fs[v] = __expf(mrow[v] - mnew);
      mrow[v] = mnew;
      float p0 = __expf(s0[v] - mnew);
      float p1 = __expf(s1[v] - mnew);
      s0[v] = p0;
      s1[v] = p1;
      float rs = p0 + p1;
#pragma unroll
      for (int m = 8; m > 0; m >>= 1) rs += __shfl_xor(rs, m, 32);
      lrow[v] = lrow[v] * fs[v] + rs;
    }
#pragma unroll
    for (int dt = 0; dt < 8; ++dt)
#pragma unroll
      for (int v = 0; v < 8; ++v) od[dt][v] *= fs[v];

    // write P (f16) to LDS in C/D layout
#pragma unroll
    for (int v = 0; v < 8; ++v) {
      Ps[wave][v + hi * 8][l16]      = (_Float16)s0[v];
      Ps[wave][v + hi * 8][16 + l16] = (_Float16)s1[v];
    }
    __syncthreads();

    // read P as A-fragment
    v16h pa;
    {
      int kb = hi * 8;
      v8h lo  = *(const v8h*)&Ps[wave][l16][kb];
      v8h hi2 = *(const v8h*)&Ps[wave][l16][kb + 16];
      pa = cat8(lo, hi2);
    }
    // O += P * V, 8 d-tiles
#pragma unroll
    for (int dt = 0; dt < 8; ++dt) {
      int dd = dt * 16 + l16;
      int kf = hi * 8;
      v8h lo  = *(const v8h*)&Vt[dd][kf];
      v8h hi2 = *(const v8h*)&Vt[dd][kf + 16];
      v16h vb = cat8(lo, hi2);
      od[dt] = WMMA_F16(pa, vb, od[dt]);
    }
  }

  // normalize and store attention output in (B*T, D) layout for the Wout GEMM
  const size_t obase = ((size_t)b * T) * D + (size_t)h * DH;
#pragma unroll
  for (int dt = 0; dt < 8; ++dt) {
    int dd = dt * 16 + l16;
#pragma unroll
    for (int v = 0; v < 8; ++v) {
      int r = q0 + v + hi * 8;
      oh[obase + (size_t)r * D + dd] = (_Float16)(od[dt][v] / lrow[v]);
    }
  }
}

// ---------------------------------------------------------------------------
// host launcher
// ---------------------------------------------------------------------------
extern "C" void kernel_launch(void* const* d_in, const int* in_sizes, int n_in,
                              void* d_out, int out_size, void* d_ws, size_t ws_size,
                              hipStream_t stream) {
  (void)in_sizes; (void)n_in; (void)out_size; (void)ws_size;
  const float* x    = (const float*)d_in[0];
  const float* mask = (const float*)d_in[1];
  const float* Wq   = (const float*)d_in[2];
  const float* Wkc  = (const float*)d_in[3];
  const float* Wvc  = (const float*)d_in[4];
  const float* Wkt  = (const float*)d_in[5];
  const float* Wvt  = (const float*)d_in[6];
  const float* Wb   = (const float*)d_in[7];
  const float* bb   = (const float*)d_in[8];
  const float* Wout = (const float*)d_in[9];
  const float* Wg1  = (const float*)d_in[10];
  const float* bg1  = (const float*)d_in[11];
  const float* Wg2  = (const float*)d_in[12];
  const float* bg2  = (const float*)d_in[13];

  constexpr int B = 2, T = 2048, D = 2048, H = 16, G = 4;
  const int BT = B * T;                 // 4096
  const size_t ND  = (size_t)D * D;     // 4M elements
  const size_t NBT = (size_t)BT * D;    // 8M elements

  char* ws = (char*)d_ws;
  size_t off = 0;
  auto alloc = [&](size_t bytes) -> char* {
    char* p = ws + off;
    off += (bytes + 255) & ~(size_t)255;
    return p;
  };
  _Float16* xh    = (_Float16*)alloc(NBT * 2);
  _Float16* Wqh   = (_Float16*)alloc(ND * 2);
  _Float16* Wkch  = (_Float16*)alloc(ND * 2);
  _Float16* Wvch  = (_Float16*)alloc(ND * 2);
  _Float16* Wkth  = (_Float16*)alloc(ND * 2);
  _Float16* Wvth  = (_Float16*)alloc(ND * 2);
  _Float16* Wouth = (_Float16*)alloc(ND * 2);
  _Float16* qhh   = (_Float16*)alloc(NBT * 2);
  _Float16* kch   = (_Float16*)alloc(NBT * 2);
  _Float16* vch   = (_Float16*)alloc(NBT * 2);
  _Float16* kth   = (_Float16*)alloc(NBT * 2);
  _Float16* vth   = (_Float16*)alloc(NBT * 2);
  _Float16* khh   = (_Float16*)alloc(NBT * 2);
  _Float16* vhh   = (_Float16*)alloc(NBT * 2);
  _Float16* ohh   = (_Float16*)alloc(NBT * 2);
  float*    blend = (float*)alloc((size_t)BT * H * 4);
  float*    ikm   = (float*)alloc(64);
  float*    gateb = (float*)alloc((size_t)BT * G * 4);

  float* out_o = (float*)d_out;   // (B,T,D)
  float* out_k = out_o + NBT;     // (B,H,T,DH)
  float* out_v = out_k + NBT;     // (B,H,T,DH)

  // 1) f32 -> f16 staging
  cvt_f16_kernel<<<(unsigned)((NBT + 255) / 256), 256, 0, stream>>>(x, xh, NBT);
  cvt_f16_kernel<<<(unsigned)((ND + 255) / 256), 256, 0, stream>>>(Wq, Wqh, ND);
  cvt_f16_kernel<<<(unsigned)((ND + 255) / 256), 256, 0, stream>>>(Wkc, Wkch, ND);
  cvt_f16_kernel<<<(unsigned)((ND + 255) / 256), 256, 0, stream>>>(Wvc, Wvch, ND);
  cvt_f16_kernel<<<(unsigned)((ND + 255) / 256), 256, 0, stream>>>(Wkt, Wkth, ND);
  cvt_f16_kernel<<<(unsigned)((ND + 255) / 256), 256, 0, stream>>>(Wvt, Wvth, ND);
  cvt_f16_kernel<<<(unsigned)((ND + 255) / 256), 256, 0, stream>>>(Wout, Wouth, ND);

  // 2) blend, mask mean, gate MLP
  blend_kernel<<<(BT * H) / 256, 256, 0, stream>>>(x, Wb, bb, blend);
  mean_kernel<<<B, 256, 0, stream>>>(mask, ikm, T);
  gate_kernel<<<(BT + 255) / 256, 256, 0, stream>>>(mask, ikm, Wg1, bg1, Wg2, bg2, gateb, T);

  // 3) projections: (BT x D) @ (D x D), f16 outputs
  dim3 ggrid(D / 128, BT / 128);
  gemm_f16_kernel<false><<<ggrid, 256, 0, stream>>>(xh, Wqh,  qhh, BT, D, D);
  gemm_f16_kernel<false><<<ggrid, 256, 0, stream>>>(xh, Wkch, kch, BT, D, D);
  gemm_f16_kernel<false><<<ggrid, 256, 0, stream>>>(xh, Wvch, vch, BT, D, D);
  gemm_f16_kernel<false><<<ggrid, 256, 0, stream>>>(xh, Wkth, kth, BT, D, D);
  gemm_f16_kernel<false><<<ggrid, 256, 0, stream>>>(xh, Wvth, vth, BT, D, D);

  // 4) k/v blend-combine: fp32 k,v into d_out + f16 copies for attention
  combine_kv_kernel<<<(unsigned)(NBT / 256), 256, 0, stream>>>(
      kch, kth, vch, vth, blend, out_k, out_v, khh, vhh);

  // 5) flash attention
  dim3 agrid(B * H, T / 128);
  attn_kernel<<<agrid, 256, 0, stream>>>(qhh, khh, vhh, gateb, ohh);

  // 6) output projection: fp32 result straight into d_out
  gemm_f16_kernel<true><<<ggrid, 256, 0, stream>>>(ohh, Wouth, out_o, BT, D, D);
}
